// FamilyGNN_75093208203877
// MI455X (gfx1250) — matrix operand order repaired and checked by reading, not compiled, not run on hardware
//
#include <hip/hip_runtime.h>
#include <hip/hip_bf16.h>

typedef __attribute__((ext_vector_type(16))) _Float16 v16h;
typedef __attribute__((ext_vector_type(8)))  float    v8f;

#define NEG_SLOPE 0.2f

// ---------- ordered-uint encoding for float atomicMax ----------
__device__ __forceinline__ unsigned enc_f32(float f) {
  unsigned u = __float_as_uint(f);
  return (u & 0x80000000u) ? ~u : (u | 0x80000000u);
}
__device__ __forceinline__ float dec_f32(unsigned k) {
  unsigned u = (k & 0x80000000u) ? (k & 0x7FFFFFFFu) : ~k;
  return __uint_as_float(u);
}

// ---------- pack fp32 weights [K,N] into per-lane WMMA B fragments ----------
// B fragment (f16, K=32, N=16): lane l -> col n = l%16, K-half = l/16,
// element e (0..15) -> K = ktile*32 + 16*(l/16) + e. 16 f16 per lane contiguous.
__global__ void pack_weights(const float* __restrict__ W, _Float16* __restrict__ out,
                             int K, int N) {
  int idx = blockIdx.x * blockDim.x + threadIdx.x;
  if (idx >= K * N) return;
  int e    = idx & 15;
  int lane = (idx >> 4) & 31;
  int tile = idx >> 9;                 // (kt * ntiles + nt)
  int ntiles = N >> 4;
  int kt = tile / ntiles;
  int nt = tile % ntiles;
  int n = nt * 16 + (lane & 15);
  int k = kt * 32 + 16 * (lane >> 4) + e;
  out[idx] = (_Float16)W[(size_t)k * N + n];
}

// ---------- raw A fragment (fp32) + conversion ----------
struct AF { float4 f0, f1, f2, f3; };

__device__ __forceinline__ AF load_afrag(const float* __restrict__ arow, int k0, int half) {
  const float4* p0 = (const float4*)(arow + k0 + 16 * half);
  const float4* p1 = (const float4*)(arow + k0 + 8 + 16 * half);
  AF r;
  r.f0 = p0[0]; r.f1 = p0[1];
  r.f2 = p1[0]; r.f3 = p1[1];
  return r;
}

__device__ __forceinline__ v16h cvt_afrag(const AF& af) {
  v16h a;
  a[0]  = (_Float16)af.f0.x; a[1]  = (_Float16)af.f0.y;
  a[2]  = (_Float16)af.f0.z; a[3]  = (_Float16)af.f0.w;
  a[4]  = (_Float16)af.f1.x; a[5]  = (_Float16)af.f1.y;
  a[6]  = (_Float16)af.f1.z; a[7]  = (_Float16)af.f1.w;
  a[8]  = (_Float16)af.f2.x; a[9]  = (_Float16)af.f2.y;
  a[10] = (_Float16)af.f2.z; a[11] = (_Float16)af.f2.w;
  a[12] = (_Float16)af.f3.x; a[13] = (_Float16)af.f3.y;
  a[14] = (_Float16)af.f3.z; a[15] = (_Float16)af.f3.w;
  return a;
}

// ---------- WMMA GEMM: C[M,N] = epilogue(A[M,K] * B + bias + addsrc) ----------
// Block = 8 waves sharing one 64-col group; B panel staged in LDS (<=32KB).
// Each wave: 16 rows x 64 cols, f16 inputs, f32 accumulate, A double-buffered
// with the last K-step peeled so the steady-state prefetch is unconditional.
__global__ __launch_bounds__(256)
void gemm_wmma(const float* __restrict__ A, const _Float16* __restrict__ Bp,
               float* __restrict__ C, const float* __restrict__ bias,
               const float* __restrict__ addsrc, int M, int K, int N, int relu) {
  __shared__ _Float16 ldsB[16384];      // up to K=256: 8 ktiles * 2048 halfs = 32KB

  int lane   = threadIdx.x & 31;
  int waveid = threadIdx.x >> 5;
  int ngroups = N >> 6;
  int mtiles  = M >> 4;
  int ng = blockIdx.x % ngroups;
  int mt = (blockIdx.x / ngroups) * 8 + waveid;
  int ntiles = N >> 4;
  int ktiles = K >> 5;

  // cooperative copy of this block's B panel (4 tiles * 512 halfs contiguous per kt)
  int total16 = (ktiles * 2048) >> 3;   // uint4 chunks
  for (int idx = threadIdx.x; idx < total16; idx += 256) {
    int kt  = idx >> 8;                 // 256 uint4 per ktile
    int off = idx & 255;
    const uint4* src = (const uint4*)(Bp + ((size_t)kt * ntiles + ng * 4) * 512) + off;
    ((uint4*)ldsB)[kt * 256 + off] = *src;
  }
  __syncthreads();

  if (mt >= mtiles) return;

  int mrow = lane & 15;
  int half = lane >> 4;
  const float* arow = A + (size_t)(mt * 16 + mrow) * K;

  v8f acc[4] = {v8f{}, v8f{}, v8f{}, v8f{}};
  AF a_cur = load_afrag(arow, 0, half);

  int k0 = 0;
  for (; k0 + 32 < K; k0 += 32) {
    AF a_nxt = load_afrag(arow, k0 + 32, half);   // unconditional prefetch
    v16h a = cvt_afrag(a_cur);
    const _Float16* lb = ldsB + (k0 >> 5) * 2048 + lane * 16;
    v16h b0 = *(const v16h*)(lb);
    v16h b1 = *(const v16h*)(lb + 512);
    v16h b2 = *(const v16h*)(lb + 1024);
    v16h b3 = *(const v16h*)(lb + 1536);
    acc[0] = __builtin_amdgcn_wmma_f32_16x16x32_f16(false, a, false, b0, (short)0, acc[0], false, false);
    acc[1] = __builtin_amdgcn_wmma_f32_16x16x32_f16(false, a, false, b1, (short)0, acc[1], false, false);
    acc[2] = __builtin_amdgcn_wmma_f32_16x16x32_f16(false, a, false, b2, (short)0, acc[2], false, false);
    acc[3] = __builtin_amdgcn_wmma_f32_16x16x32_f16(false, a, false, b3, (short)0, acc[3], false, false);
    a_cur = a_nxt;
  }
  {                                                // peeled final K-step
    v16h a = cvt_afrag(a_cur);
    const _Float16* lb = ldsB + (k0 >> 5) * 2048 + lane * 16;
    v16h b0 = *(const v16h*)(lb);
    v16h b1 = *(const v16h*)(lb + 512);
    v16h b2 = *(const v16h*)(lb + 1024);
    v16h b3 = *(const v16h*)(lb + 1536);
    acc[0] = __builtin_amdgcn_wmma_f32_16x16x32_f16(false, a, false, b0, (short)0, acc[0], false, false);
    acc[1] = __builtin_amdgcn_wmma_f32_16x16x32_f16(false, a, false, b1, (short)0, acc[1], false, false);
    acc[2] = __builtin_amdgcn_wmma_f32_16x16x32_f16(false, a, false, b2, (short)0, acc[2], false, false);
    acc[3] = __builtin_amdgcn_wmma_f32_16x16x32_f16(false, a, false, b3, (short)0, acc[3], false, false);
  }

  // epilogue: D layout -> vgpr v: row = mt*16 + v + 8*half, col = subtile + lane%16
#pragma unroll
  for (int t = 0; t < 4; ++t) {
    int n = ng * 64 + t * 16 + (lane & 15);
    float bv = bias ? bias[n] : 0.0f;
#pragma unroll
    for (int v = 0; v < 8; ++v) {
      int row = mt * 16 + v + 8 * half;
      float val = acc[t][v] + bv;
      if (addsrc) val += addsrc[(size_t)row * N + n];
      if (relu) val = fmaxf(val, 0.0f);
      C[(size_t)row * N + n] = val;
    }
  }
}

// ---------- per-node attention coefficients a_src/a_dst [N,H] ----------
__global__ void node_att(const float* __restrict__ hW, const float* __restrict__ att_s,
                         const float* __restrict__ att_d, float* __restrict__ a_s,
                         float* __restrict__ a_d, int Nn, int H, int Cc) {
  int idx = blockIdx.x * blockDim.x + threadIdx.x;
  if (idx >= Nn * H) return;
  int n = idx / H, h = idx % H;
  const float* p = hW + (size_t)n * H * Cc + (size_t)h * Cc;
  float s = 0.f, d = 0.f;
  for (int c = 0; c < Cc; ++c) {
    float v = p[c];
    s += v * att_s[h * Cc + c];
    d += v * att_d[h * Cc + c];
  }
  a_s[idx] = s;
  a_d[idx] = d;
}

__global__ void init_nodes(unsigned* __restrict__ nmax, float* __restrict__ nsum, int count) {
  int i = blockIdx.x * blockDim.x + threadIdx.x;
  if (i < count) { nmax[i] = 0x007FFFFFu; /* enc(-inf) */ nsum[i] = 0.0f; }
}

// ---------- edge pass 1: segment max of leaky_relu logits ----------
__global__ void edge_max(const int* __restrict__ src, const int* __restrict__ dst,
                         const float* __restrict__ a_s, const float* __restrict__ a_d,
                         unsigned* __restrict__ nmax, int E0, int Nn, int H) {
  int e = blockIdx.x * blockDim.x + threadIdx.x;
  if (e >= E0 + Nn) return;
  int s = (e < E0) ? src[e] : (e - E0);
  int d = (e < E0) ? dst[e] : (e - E0);
  for (int h = 0; h < H; ++h) {
    float v = a_s[s * H + h] + a_d[d * H + h];
    v = (v > 0.f) ? v : NEG_SLOPE * v;
    atomicMax(&nmax[d * H + h], enc_f32(v));
  }
}

// ---------- edge pass 2: segment sum of exp(e - max) ----------
__global__ void edge_sum(const int* __restrict__ src, const int* __restrict__ dst,
                         const float* __restrict__ a_s, const float* __restrict__ a_d,
                         const unsigned* __restrict__ nmax, float* __restrict__ nsum,
                         int E0, int Nn, int H) {
  int e = blockIdx.x * blockDim.x + threadIdx.x;
  if (e >= E0 + Nn) return;
  int s = (e < E0) ? src[e] : (e - E0);
  int d = (e < E0) ? dst[e] : (e - E0);
  for (int h = 0; h < H; ++h) {
    float v = a_s[s * H + h] + a_d[d * H + h];
    v = (v > 0.f) ? v : NEG_SLOPE * v;
    float m = dec_f32(nmax[d * H + h]);
    atomicAdd(&nsum[d * H + h], expf(v - m));
  }
}

// ---------- edge pass 3: out[dst] += alpha * hW[src], 32 channels / thread ----------
// hW chunk read as 8x float4 (b128); atomics stay per-f32 (L2-resident).
__global__ void edge_scatter(const int* __restrict__ src, const int* __restrict__ dst,
                             const float* __restrict__ a_s, const float* __restrict__ a_d,
                             const unsigned* __restrict__ nmax, const float* __restrict__ nsum,
                             const float* __restrict__ hW, float* __restrict__ out,
                             int E0, int Nn, int H, int Cc) {
  int D = H * Cc;
  int chunks = D >> 5;
  int t = blockIdx.x * blockDim.x + threadIdx.x;
  int e = t / chunks;
  int ch = t % chunks;
  if (e >= E0 + Nn) return;
  int s = (e < E0) ? src[e] : (e - E0);
  int d = (e < E0) ? dst[e] : (e - E0);
  int c0 = ch * 32;
  int h = c0 / Cc;                       // 32-chunk always within one head (Cc >= 32)
  float v = a_s[s * H + h] + a_d[d * H + h];
  v = (v > 0.f) ? v : NEG_SLOPE * v;
  float m = dec_f32(nmax[d * H + h]);
  float alpha = expf(v - m) / (nsum[d * H + h] + 1e-16f);
  const float4* hp = (const float4*)(hW + (size_t)s * D + c0);
  float* op = out + (size_t)d * D + c0;
#pragma unroll
  for (int j = 0; j < 8; ++j) {
    float4 hv = hp[j];
    atomicAdd(&op[j * 4 + 0], alpha * hv.x);
    atomicAdd(&op[j * 4 + 1], alpha * hv.y);
    atomicAdd(&op[j * 4 + 2], alpha * hv.z);
    atomicAdd(&op[j * 4 + 3], alpha * hv.w);
  }
}

// ---------- fused bias + LayerNorm + optional ReLU, one wave32 per node ----------
__global__ __launch_bounds__(256)
void layernorm_k(float* __restrict__ hio, const float* __restrict__ bias,
                 const float* __restrict__ g, const float* __restrict__ b,
                 int Nn, int D, int relu) {
  int wave = (blockIdx.x * blockDim.x + threadIdx.x) >> 5;
  int lane = threadIdx.x & 31;
  if (wave >= Nn) return;
  float* p = hio + (size_t)wave * D;
  int per = D >> 5;                     // 8 (D=256) or 4 (D=128)
  float vals[8];
  float sum = 0.f;
  for (int j = 0; j < per; ++j) {
    int c = lane + j * 32;
    float v = p[c];
    if (bias) v += bias[c];
    vals[j] = v;
    sum += v;
  }
  for (int o = 16; o > 0; o >>= 1) sum += __shfl_xor(sum, o, 32);
  float mu = sum / (float)D;
  float var = 0.f;
  for (int j = 0; j < per; ++j) { float dv = vals[j] - mu; var += dv * dv; }
  for (int o = 16; o > 0; o >>= 1) var += __shfl_xor(var, o, 32);
  var /= (float)D;
  float inv = rsqrtf(var + 1e-5f);
  for (int j = 0; j < per; ++j) {
    int c = lane + j * 32;
    float v = (vals[j] - mu) * inv * g[c] + b[c];
    if (relu) v = fmaxf(v, 0.f);
    p[c] = v;
  }
}

// =======================================================================
extern "C" void kernel_launch(void* const* d_in, const int* in_sizes, int n_in,
                              void* d_out, int out_size, void* d_ws, size_t ws_size,
                              hipStream_t stream) {
  const int Nn = 50000, IN_DIM = 128, HID = 256, OUT_DIM = 128;
  const float* x      = (const float*)d_in[0];
  const int*   eidx   = (const int*)d_in[1];
  const float* in_W   = (const float*)d_in[2];
  const float* in_b   = (const float*)d_in[3];
  const float* skip_W = (const float*)d_in[4];
  const float* skip_b = (const float*)d_in[5];
  const int E0 = in_sizes[1] / 2;
  const int* e_src = eidx;
  const int* e_dst = eidx + E0;

  // ---- workspace carve (all offsets 256B aligned) ----
  char* ws = (char*)d_ws;
  const size_t FEAT = (size_t)Nn * HID * sizeof(float);       // 51.2 MB
  const size_t NODE8 = (size_t)Nn * 8 * sizeof(float);        // 1.6 MB
  float*    bufA  = (float*)(ws);                             // h / aggregation
  float*    bufB  = (float*)(ws + FEAT);                      // hW
  float*    hinit = (float*)(ws + 2 * FEAT);
  float*    a_s   = (float*)(ws + 3 * FEAT);
  float*    a_d   = (float*)(ws + 3 * FEAT + NODE8);
  unsigned* nmax  = (unsigned*)(ws + 3 * FEAT + 2 * NODE8);
  float*    nsum  = (float*)(ws + 3 * FEAT + 3 * NODE8);
  _Float16* pw    = (_Float16*)(ws + 3 * FEAT + 4 * NODE8);
  _Float16* inWp   = pw;                         // 128*256
  _Float16* Wp0    = pw + 32768;                 // 256*256
  _Float16* Wp1    = pw + 32768 + 65536;
  _Float16* Wp2    = pw + 32768 + 2 * 65536;
  _Float16* Wp3    = pw + 32768 + 3 * 65536;     // 256*128
  _Float16* skipWp = pw + 32768 + 3 * 65536 + 32768;

  // ---- pack weights into WMMA B-fragment layout ----
  auto packgrid = [](int kn) { return (kn + 255) / 256; };
  pack_weights<<<packgrid(IN_DIM * HID), 256, 0, stream>>>(in_W, inWp, IN_DIM, HID);
  pack_weights<<<packgrid(HID * HID), 256, 0, stream>>>((const float*)d_in[6],  Wp0, HID, HID);
  pack_weights<<<packgrid(HID * HID), 256, 0, stream>>>((const float*)d_in[12], Wp1, HID, HID);
  pack_weights<<<packgrid(HID * HID), 256, 0, stream>>>((const float*)d_in[18], Wp2, HID, HID);
  pack_weights<<<packgrid(HID * OUT_DIM), 256, 0, stream>>>((const float*)d_in[24], Wp3, HID, OUT_DIM);
  pack_weights<<<packgrid(HID * OUT_DIM), 256, 0, stream>>>(skip_W, skipWp, HID, OUT_DIM);

  // grid: each block = 8 M-tiles x one 64-col group (shared LDS B panel)
  auto gemm_blocks = [&](int M, int N) {
    int ngroups = N >> 6;
    int mblocks = ((M >> 4) + 7) / 8;
    return ngroups * mblocks;
  };

  // ---- input projection: h0 = relu(x @ in_W + in_b) ----
  gemm_wmma<<<gemm_blocks(Nn, HID), 256, 0, stream>>>(x, inWp, bufA, in_b, nullptr,
                                                      Nn, IN_DIM, HID, 1);
  hipMemcpyAsync(hinit, bufA, FEAT, hipMemcpyDeviceToDevice, stream);

  const int Et = E0 + Nn;
  _Float16* Wps[4] = {Wp0, Wp1, Wp2, Wp3};

  for (int i = 0; i < 4; ++i) {
    int base = 6 + 6 * i;
    const float* att_src_p = (const float*)d_in[base + 1];
    const float* att_dst_p = (const float*)d_in[base + 2];
    const float* bias_p    = (const float*)d_in[base + 3];
    const float* ln_g      = (const float*)d_in[base + 4];
    const float* ln_b      = (const float*)d_in[base + 5];
    int H = (i < 3) ? 8 : 1;
    int C = (i < 3) ? 32 : 128;
    int D = H * C;

    // hW = h @ W   (bufA -> bufB)
    gemm_wmma<<<gemm_blocks(Nn, D), 256, 0, stream>>>(bufA, Wps[i], bufB, nullptr, nullptr,
                                                      Nn, HID, D, 0);
    // attention coefficients
    node_att<<<(Nn * H + 255) / 256, 256, 0, stream>>>(bufB, att_src_p, att_dst_p,
                                                       a_s, a_d, Nn, H, C);
    init_nodes<<<(Nn * H + 255) / 256, 256, 0, stream>>>(nmax, nsum, Nn * H);
    hipMemsetAsync(bufA, 0, (size_t)Nn * D * sizeof(float), stream);

    // segment softmax + weighted aggregation (self-loops implicit at e >= E0)
    edge_max<<<(Et + 255) / 256, 256, 0, stream>>>(e_src, e_dst, a_s, a_d, nmax, E0, Nn, H);
    edge_sum<<<(Et + 255) / 256, 256, 0, stream>>>(e_src, e_dst, a_s, a_d, nmax, nsum, E0, Nn, H);
    int sc_threads = Et * (D >> 5);
    edge_scatter<<<(sc_threads + 255) / 256, 256, 0, stream>>>(e_src, e_dst, a_s, a_d, nmax,
                                                               nsum, bufB, bufA, E0, Nn, H, C);
    // bias + LayerNorm (+ ReLU on layers 0..2), in place on bufA
    layernorm_k<<<(Nn * 32 + 255) / 256, 256, 0, stream>>>(bufA, bias_p, ln_g, ln_b,
                                                           Nn, D, (i < 3) ? 1 : 0);
  }

  // ---- final skip: out = h3 + h_init @ skip_W + skip_b ----
  gemm_wmma<<<gemm_blocks(Nn, OUT_DIM), 256, 0, stream>>>(hinit, skipWp, (float*)d_out,
                                                          skip_b, bufA, Nn, HID, OUT_DIM, 0);
}
